// spiking_transformer_86260123173262
// MI455X (gfx1250) — matrix operand order repaired and checked by reading, not compiled, not run on hardware
//
#include <hip/hip_runtime.h>

#define T_ 4
#define B_ 64
#define C_ 256
#define N_ 256
#define BN_EPS 1e-5f

typedef __attribute__((ext_vector_type(16))) _Float16 v16h;
typedef __attribute__((ext_vector_type(8)))  _Float16 v8h;
typedef __attribute__((ext_vector_type(8)))  float    v8f;

// ---------------------------------------------------------------------------
// A-fragment (16x32 f16, M x K) for wave32 WMMA:
//   lanes 0-15 : M = lane,     halves 0..15 = K[0..15]
//   lanes 16-31: M = lane-16,  halves 0..15 = K[16..31]
// Row-major f16 weights [O][C]: two aligned 16-byte loads per lane.
__device__ __forceinline__ v16h load_afrag(const _Float16* __restrict__ w,
                                           int row, int k0, int hi) {
  const _Float16* p = w + row * C_ + k0 + hi * 16;
  v16h a;
  ((v8h*)&a)[0] = *(const v8h*)p;
  ((v8h*)&a)[1] = *(const v8h*)(p + 8);
  return a;
}

// B-fragment (32x16 f16, K x N):
//   lanes 0-15 : N = lane,     halves 0..15 = K[0..15]
//   lanes 16-31: N = lane-16,  halves 0..15 = K[16..31]
// Source stored column-major-in-K (N-major): 16 contiguous halves per lane.
__device__ __forceinline__ v16h load_bfrag(const _Float16* __restrict__ p) {
  v16h a;
  ((v8h*)&a)[0] = *(const v8h*)p;
  ((v8h*)&a)[1] = *(const v8h*)(p + 8);
  return a;
}

// ---------------------------------------------------------------------------
// f32 -> f16 weight conversion (row-major preserved; weights stay L2-resident)
__global__ void cvt_f32_f16(const float* __restrict__ s, _Float16* __restrict__ d, int n) {
  int i = blockIdx.x * 256 + threadIdx.x;
  if (i < n) d[i] = (_Float16)s[i];
}

// ---------------------------------------------------------------------------
// LIF over the raw input, producing spikes in N-major f16 layout sT[t][b][n][c]
// (the layout WMMA B-fragments want). 32x32 tile transpose through LDS so both
// the f32 reads and the f16 writes are coalesced; membranes persist in regs
// across the T loop.
__global__ __launch_bounds__(256)
void lif_input(const float* __restrict__ x, _Float16* __restrict__ sT) {
  int blk = blockIdx.x;              // B * (C/32) * (N/32) = 64*8*8
  int b   = blk >> 6;
  int rs  = blk & 63;
  int c0  = (rs >> 3) << 5;
  int n0  = (rs & 7) << 5;
  int tx  = threadIdx.x & 31;
  int ty  = threadIdx.x >> 5;        // 8 rows per pass

  __shared__ float tile[32][33];
  float mem[4] = {0.f, 0.f, 0.f, 0.f};

  for (int t = 0; t < T_; ++t) {
    #pragma unroll
    for (int j = 0; j < 4; ++j) {
      int c = c0 + ty + j * 8;
      float xv = x[(((size_t)t * B_ + b) * C_ + c) * N_ + n0 + tx];
      float mm = mem[j] * 0.5f + xv;
      float sp = (mm >= 1.0f) ? 1.0f : 0.0f;
      mem[j] = mm * (1.0f - sp);
      tile[ty + j * 8][tx] = sp;
    }
    __syncthreads();
    #pragma unroll
    for (int j = 0; j < 4; ++j) {
      int n = n0 + ty + j * 8;
      sT[(((size_t)t * B_ + b) * N_ + n) * C_ + c0 + tx] = (_Float16)tile[tx][ty + j * 8];
    }
    __syncthreads();
  }
}

// ---------------------------------------------------------------------------
// Fused q/k GEMM + BN + LIF + decay memory + head-sum + attn-LIF + y=attn*k
// + proj GEMM + bias + BN.  One workgroup = (batch b, 16-column N tile).
// 16 waves; wave m owns output-channel tile [16m, 16m+16).  All temporal state
// (q/k membranes, decay mems, attn membrane) lives in registers across t.
// The spike tile for the 16 columns is staged once per t into LDS and shared
// by all 16 waves (both q and k GEMMs), so each B-fragment is loaded from
// global exactly once per block.
__global__ __launch_bounds__(512)
void attn_fused(const _Float16* __restrict__ sT,
                const _Float16* __restrict__ qw,
                const _Float16* __restrict__ kw,
                const _Float16* __restrict__ pw,
                const float* __restrict__ q_gamma, const float* __restrict__ q_beta,
                const float* __restrict__ q_mean,  const float* __restrict__ q_var,
                const float* __restrict__ k_gamma, const float* __restrict__ k_beta,
                const float* __restrict__ k_mean,  const float* __restrict__ k_var,
                const float* __restrict__ p_bias,
                const float* __restrict__ p_gamma, const float* __restrict__ p_beta,
                const float* __restrict__ p_mean,  const float* __restrict__ p_var,
                const float* __restrict__ alpha_p,
                float* __restrict__ out) {
  __shared__ __align__(32) _Float16 sTile[16][C_];  // s tile, N-major (8 KB)
  __shared__ __align__(32) _Float16 yT[16][C_];     // y tile, N-major (8 KB)
  __shared__ float qsum_s[8][2][16];                // per-head partial sums

  const int b    = blockIdx.x >> 4;
  const int n0   = (blockIdx.x & 15) << 4;
  const int wv   = threadIdx.x >> 5;    // M-tile / wave id (0..15)
  const int ln   = threadIdx.x & 31;
  const int col  = ln & 15;             // N within tile (B/D lanes)
  const int hi   = ln >> 4;             // K-half select
  const int head = wv >> 1;
  const int par  = wv & 1;
  const int arow = wv * 16 + col;       // A-matrix row for this lane

  // staging assignment: thread -> (column, 8-half chunk) of the s tile
  const int scol   = threadIdx.x >> 5;  // 0..15
  const int schunk = threadIdx.x & 31;  // 0..31 (8 halves each)

  // Fold BN (and proj bias) into per-accumulator-register scale/offset.
  float qsc[8], qof[8], ksc[8], kof[8], psc[8], pof[8];
  #pragma unroll
  for (int r = 0; r < 8; ++r) {
    int c = wv * 16 + hi * 8 + r;       // D-matrix channel of accumulator reg r
    float qi = q_gamma[c] * rsqrtf(q_var[c] + BN_EPS);
    qsc[r] = qi; qof[r] = q_beta[c] - q_mean[c] * qi;
    float ki = k_gamma[c] * rsqrtf(k_var[c] + BN_EPS);
    ksc[r] = ki; kof[r] = k_beta[c] - k_mean[c] * ki;
    float pi = p_gamma[c] * rsqrtf(p_var[c] + BN_EPS);
    psc[r] = pi; pof[r] = p_bias[c] * pi + p_beta[c] - p_mean[c] * pi;
  }
  const float alpha = alpha_p[0];
  const float oma   = 1.0f - alpha;

  v8f q_mem = {0.f,0.f,0.f,0.f,0.f,0.f,0.f,0.f};
  v8f k_mem = {0.f,0.f,0.f,0.f,0.f,0.f,0.f,0.f};
  v8f mems  = {0.f,0.f,0.f,0.f,0.f,0.f,0.f,0.f};
  float attn_mem = 0.0f;

  const _Float16* stage_src =
      sT + (((size_t)b) * N_ + (n0 + scol)) * C_ + schunk * 8;

  for (int t = 0; t < T_; ++t) {
    // ---- stage spike tile into LDS (512 threads x 16 B = 8 KB) -----------
    *(v8h*)&sTile[scol][schunk * 8] =
        *(const v8h*)(stage_src + (size_t)t * B_ * N_ * C_);
    __syncthreads();

    // ---- q = qw @ s, k = kw @ s (16x16 tile, K = 256) --------------------
    v8f qa = {0.f,0.f,0.f,0.f,0.f,0.f,0.f,0.f};
    v8f ka = {0.f,0.f,0.f,0.f,0.f,0.f,0.f,0.f};
    #pragma unroll 2
    for (int ks = 0; ks < 8; ++ks) {
      v16h bf = load_bfrag(&sTile[col][ks * 32 + hi * 16]);
      v16h aq = load_afrag(qw, arow, ks * 32, hi);
      qa = __builtin_amdgcn_wmma_f32_16x16x32_f16(false, aq, false, bf,
                                                  (short)0, qa, false, false);
      v16h ak = load_afrag(kw, arow, ks * 32, hi);
      ka = __builtin_amdgcn_wmma_f32_16x16x32_f16(false, ak, false, bf,
                                                  (short)0, ka, false, false);
    }

    // ---- BN + LIF for q,k ; decay memory ; per-lane head-sum partial -----
    v8f kspk;
    float part = 0.0f;
    #pragma unroll
    for (int r = 0; r < 8; ++r) {
      float qv = qa[r] * qsc[r] + qof[r];
      float qm = q_mem[r] * 0.5f + qv;
      float sp = (qm >= 1.0f) ? 1.0f : 0.0f;
      q_mem[r] = qm * (1.0f - sp);

      float kv = ka[r] * ksc[r] + kof[r];
      float km = k_mem[r] * 0.5f + kv;
      float kp = (km >= 1.0f) ? 1.0f : 0.0f;
      k_mem[r] = km * (1.0f - kp);
      kspk[r] = kp;

      float mm = oma * mems[r] + alpha * sp;   // Wei recurrence
      mems[r] = mm;
      part += mm + sp;                         // sum over d of (mems + q)
    }
    // fold rows M=0..7 (lanes 0-15) with rows M=8..15 (lanes 16-31)
    part += __shfl_xor(part, 16, 32);

    __syncthreads();                 // prior users of qsum_s / yT are done
    if (ln < 16) qsum_s[head][par][ln] = part;
    __syncthreads();
    float qsum = qsum_s[head][0][col] + qsum_s[head][1][col];

    // ---- attn LIF (threshold 0.5), state per (head, n) -------------------
    float am  = attn_mem * 0.5f + qsum;
    float asp = (am >= 0.5f) ? 1.0f : 0.0f;
    attn_mem = am * (1.0f - asp);

    // ---- y = attn * k -> LDS in N-major f16 (B-fragment layout) ----------
    v8h yh;
    #pragma unroll
    for (int r = 0; r < 8; ++r) yh[r] = (_Float16)(asp * kspk[r]);
    *(v8h*)&yT[col][wv * 16 + hi * 8] = yh;
    __syncthreads();

    // ---- proj GEMM from LDS + bias + BN ----------------------------------
    v8f pa = {0.f,0.f,0.f,0.f,0.f,0.f,0.f,0.f};
    #pragma unroll 2
    for (int ks = 0; ks < 8; ++ks) {
      v16h bp = load_bfrag(&yT[col][ks * 32 + hi * 16]);
      v16h ap = load_afrag(pw, arow, ks * 32, hi);
      pa = __builtin_amdgcn_wmma_f32_16x16x32_f16(false, ap, false, bp,
                                                  (short)0, pa, false, false);
    }
    #pragma unroll
    for (int r = 0; r < 8; ++r) {
      int c = wv * 16 + hi * 8 + r;
      out[(((size_t)t * B_ + b) * C_ + c) * N_ + n0 + col] = pa[r] * psc[r] + pof[r];
    }
  }
}

// ---------------------------------------------------------------------------
extern "C" void kernel_launch(void* const* d_in, const int* in_sizes, int n_in,
                              void* d_out, int out_size, void* d_ws, size_t ws_size,
                              hipStream_t stream) {
  (void)in_sizes; (void)n_in; (void)out_size; (void)ws_size;

  const float* x       = (const float*)d_in[0];
  const float* q_w     = (const float*)d_in[1];
  const float* q_gamma = (const float*)d_in[2];
  const float* q_beta  = (const float*)d_in[3];
  const float* q_mean  = (const float*)d_in[4];
  const float* q_var   = (const float*)d_in[5];
  const float* k_w     = (const float*)d_in[6];
  const float* k_gamma = (const float*)d_in[7];
  const float* k_beta  = (const float*)d_in[8];
  const float* k_mean  = (const float*)d_in[9];
  const float* k_var   = (const float*)d_in[10];
  const float* proj_w  = (const float*)d_in[11];
  const float* proj_b  = (const float*)d_in[12];
  const float* p_gamma = (const float*)d_in[13];
  const float* p_beta  = (const float*)d_in[14];
  const float* p_mean  = (const float*)d_in[15];
  const float* p_var   = (const float*)d_in[16];
  const float* alpha   = (const float*)d_in[17];
  float* out = (float*)d_out;

  // workspace layout
  char* ws = (char*)d_ws;
  _Float16* sT   = (_Float16*)ws;                       // T*B*N*C f16 = 32 MiB
  _Float16* qw16 = (_Float16*)(ws + (size_t)T_ * B_ * N_ * C_ * 2);
  _Float16* kw16 = qw16 + C_ * C_;
  _Float16* pw16 = kw16 + C_ * C_;

  cvt_f32_f16<<<(C_ * C_ + 255) / 256, 256, 0, stream>>>(q_w, qw16, C_ * C_);
  cvt_f32_f16<<<(C_ * C_ + 255) / 256, 256, 0, stream>>>(k_w, kw16, C_ * C_);
  cvt_f32_f16<<<(C_ * C_ + 255) / 256, 256, 0, stream>>>(proj_w, pw16, C_ * C_);

  lif_input<<<B_ * (C_ / 32) * (N_ / 32), 256, 0, stream>>>(x, sT);

  attn_fused<<<B_ * (N_ / 16), 512, 0, stream>>>(
      sT, qw16, kw16, pw16,
      q_gamma, q_beta, q_mean, q_var,
      k_gamma, k_beta, k_mean, k_var,
      proj_b, p_gamma, p_beta, p_mean, p_var,
      alpha, out);
}
